// WSAD_82652350644503
// MI455X (gfx1250) — matrix operand order repaired and checked by reading, not compiled, not run on hardware
//
#include <hip/hip_runtime.h>
#include <hip/hip_bf16.h>

// ---------------------------------------------------------------------------
// Types
// ---------------------------------------------------------------------------
typedef __attribute__((ext_vector_type(16))) __bf16 bf16x16;
typedef __attribute__((ext_vector_type(8)))  float  floatx8;
typedef __attribute__((ext_vector_type(4)))  unsigned int uint32x4;

#define HEADS 4
#define DH    128
#define DEPTH 2
#define CDIM  512
#define TLEN  256
#define BNB   160          // 16 * 10
#define MROWS 40960        // BNB * TLEN
#define EPSLN 1e-5f

// CDNA5 async global->LDS copy (ASYNCcnt-tracked DMA), 16 bytes per lane.
__device__ __forceinline__ void async_b128(const __bf16* g, void* lds) {
  unsigned l = (unsigned)(unsigned long long)lds;   // low 32 bits = LDS offset
  asm volatile("global_load_async_to_lds_b128 %0, %1, off"
               :: "v"(l), "v"(g) : "memory");
}
__device__ __forceinline__ void wait_async0() {
  asm volatile("s_wait_asynccnt 0" ::: "memory");
}

// ---------------------------------------------------------------------------
// Elementwise: fp32 -> bf16 cast
// ---------------------------------------------------------------------------
__global__ __launch_bounds__(256) void cast_f32_bf16(const float* __restrict__ in,
                                                     __bf16* __restrict__ out, long n) {
  long i = (long)blockIdx.x * 256 + threadIdx.x;
  if (i < n) out[i] = (__bf16)in[i];
}

// ---------------------------------------------------------------------------
// Weight cast + transpose: in (L,K,N) fp32 -> out (L,N,K) bf16
// ---------------------------------------------------------------------------
__global__ __launch_bounds__(256) void cast_transpose(const float* __restrict__ in,
                                                      __bf16* __restrict__ out,
                                                      int K, int N, long total) {
  long idx = (long)blockIdx.x * 256 + threadIdx.x;
  if (idx >= total) return;
  long kn = (long)K * N;
  long l  = idx / kn;
  long r  = idx - l * kn;
  int  n  = (int)(r % N);
  int  k  = (int)(r / N);
  out[l * kn + (long)n * K + k] = (__bf16)in[idx];
}

// ---------------------------------------------------------------------------
// W_emb (512,1024,3) -> BembT[o][k=s*1024+i] bf16  ([N,K] im2col weights)
// ---------------------------------------------------------------------------
__global__ __launch_bounds__(256) void wemb_transpose(const float* __restrict__ W,
                                                      __bf16* __restrict__ out) {
  long t = (long)blockIdx.x * 256 + threadIdx.x;
  if (t >= 3072l * 512) return;
  int  o = (int)(t / 3072);
  int  k = (int)(t - (long)o * 3072);
  int  s = k >> 10;
  int  i = k & 1023;
  out[(long)o * 3072 + k] = (__bf16)W[(long)o * 3072 + i * 3 + s];
}

// ---------------------------------------------------------------------------
// a2 matrix: a2n[i][j] = exp(-|i-j|/e) / sum_k exp(-|j-k|/e)   (bf16)
// ---------------------------------------------------------------------------
__global__ __launch_bounds__(256) void a2_kernel(__bf16* __restrict__ a2) {
  int j = blockIdx.x, i = threadIdx.x;
  float e = __expf(-fabsf((float)(i - j)) * 0.36787944117144233f);  // 1/e
  __shared__ float red[256];
  red[i] = e; __syncthreads();
  for (int o = 128; o > 0; o >>= 1) { if (i < o) red[i] += red[i + o]; __syncthreads(); }
  float s = red[0];
  a2[(long)i * 256 + j] = (__bf16)(e / s);
}

// ---------------------------------------------------------------------------
// LayerNorm over C=512, one block per row, writes bf16
// ---------------------------------------------------------------------------
__global__ __launch_bounds__(256) void ln_kernel(const float* __restrict__ h,
                                                 const float* __restrict__ g,
                                                 const float* __restrict__ b,
                                                 __bf16* __restrict__ z) {
  long row = blockIdx.x;
  const float* hr = h + row * CDIM;
  int tid = threadIdx.x;
  __shared__ float red[256];
  float v0 = hr[tid], v1 = hr[tid + 256];
  red[tid] = v0 + v1; __syncthreads();
  for (int o = 128; o > 0; o >>= 1) { if (tid < o) red[tid] += red[tid + o]; __syncthreads(); }
  float mean = red[0] * (1.0f / CDIM); __syncthreads();
  float d0 = v0 - mean, d1 = v1 - mean;
  red[tid] = d0 * d0 + d1 * d1; __syncthreads();
  for (int o = 128; o > 0; o >>= 1) { if (tid < o) red[tid] += red[tid + o]; __syncthreads(); }
  float rstd = rsqrtf(red[0] * (1.0f / CDIM) + EPSLN);
  z[row * CDIM + tid]       = (__bf16)(d0 * rstd * g[tid]       + b[tid]);
  z[row * CDIM + tid + 256] = (__bf16)(d1 * rstd * g[tid + 256] + b[tid + 256]);
}

// ---------------------------------------------------------------------------
// In-place bf16 row softmax, row length 256, one block per row
// ---------------------------------------------------------------------------
__global__ __launch_bounds__(256) void softmax256(__bf16* __restrict__ S) {
  long row = blockIdx.x;
  __bf16* p = S + row * 256;
  int tid = threadIdx.x;
  float v = (float)p[tid];
  __shared__ float red[256];
  red[tid] = v; __syncthreads();
  for (int o = 128; o > 0; o >>= 1) { if (tid < o) red[tid] = fmaxf(red[tid], red[tid + o]); __syncthreads(); }
  float m = red[0]; __syncthreads();
  float e = __expf(v - m);
  red[tid] = e; __syncthreads();
  for (int o = 128; o > 0; o >>= 1) { if (tid < o) red[tid] += red[tid + o]; __syncthreads(); }
  p[tid] = (__bf16)(e / red[0]);
}

// ---------------------------------------------------------------------------
// Templated batched WMMA GEMM (bf16 A,B -> fp32 acc), double-buffered LDS with
// CDNA5 async global->LDS staging where the copy is layout-preserving:
//   - A tile (non-CONV): async;  CONV A tile: register path (zero padding)
//   - B tile (TRANSB, [N,K]): async; non-TRANSB ([K,N]): register transpose
// Block tile 128(M) x 64(N), 8 waves, each wave 16x64 -> 4 WMMAs per K-step.
// Schedule per K-step: s_wait_asynccnt 0 -> barrier -> issue next-tile async
// -> WMMAs from current buffer -> register-path stores to next buffer.
// ---------------------------------------------------------------------------
template <int TRANSB, int CONV, int ACT, int OUTBF16, int HASBIAS, int HASRES>
__global__ __launch_bounds__(256) void wmma_gemm_t(
    const __bf16* __restrict__ A, const __bf16* __restrict__ B, void* __restrict__ C,
    const float* __restrict__ bias, const float* __restrict__ resid,
    int K, int lda, int ldb, int ldc,
    long sA, long sA2, long sB, long sB2, long sC, long sC2, int bdiv,
    float alpha, int convT) {
  __shared__ __align__(16) __bf16 As[2][128][48];
  __shared__ __align__(16) __bf16 Bs[2][64][48];
  const int tid  = threadIdx.x;
  const int wave = tid >> 5;
  const int lane = tid & 31;
  const int z    = blockIdx.z;
  const int zq = z / bdiv, zr = z - zq * bdiv;
  const __bf16* Ab = A + zq * sA + zr * sA2;
  const __bf16* Bb = B + zq * sB + zr * sB2;
  const long   cOff = zq * sC + zr * sC2;
  const int m0  = blockIdx.x * 128;
  const int n0g = blockIdx.y * 64;

  const int wm = wave * 16;
  floatx8 acc0 = {}, acc1 = {}, acc2 = {}, acc3 = {};

  // staging thread mapping
  const int arow  = tid >> 1;          // 0..127
  const int acol  = (tid & 1) * 16;    // 0 or 16
  const int brow  = tid >> 2;          // 0..63   (TRANSB path)
  const int bkk   = (tid & 3) * 8;     // 0,8,16,24
  const int bkrow = tid >> 3;          // 0..31   (register path)
  const int bnn   = (tid & 7) * 8;     // 0..56

  uint32x4 aReg0, aReg1, bReg;
  const uint32x4 zz = {0u, 0u, 0u, 0u};

  // ---- async staging (layout-preserving copies) ----
  auto issueAsync = [&](int kb, int buf) {
    if (!CONV) {
      const __bf16* p = Ab + (long)(m0 + arow) * lda + kb + acol;
      async_b128(p,     &As[buf][arow][acol]);
      async_b128(p + 8, &As[buf][arow][acol + 8]);
    }
    if (TRANSB) {
      const __bf16* p = Bb + (long)(n0g + brow) * ldb + kb + bkk;
      async_b128(p, &Bs[buf][brow][bkk]);
    }
  };
  // ---- register staging (padding / transpose paths) ----
  auto regLoad = [&](int kb) {
    if (CONV) {
      int s  = kb >> 10;
      int i0 = kb & 1023;
      int srow = m0 + arow + s - 1;
      bool ok = (srow >= 0 && srow < convT);
      const __bf16* p = Ab + (long)srow * lda + i0 + acol;
      aReg0 = ok ? *(const uint32x4*)p : zz;
      aReg1 = ok ? *(const uint32x4*)(p + 8) : zz;
    }
    if (!TRANSB) {
      bReg = *(const uint32x4*)(Bb + (long)(kb + bkrow) * ldb + n0g + bnn);
    }
  };
  auto regStore = [&](int buf) {
    if (CONV) {
      *(uint32x4*)&As[buf][arow][acol]     = aReg0;
      *(uint32x4*)&As[buf][arow][acol + 8] = aReg1;
    }
    if (!TRANSB) {
      const __bf16* pv = (const __bf16*)&bReg;
#pragma unroll
      for (int e = 0; e < 8; e++) Bs[buf][bnn + e][bkrow] = pv[e];
    }
  };

  // prologue: tile 0
  issueAsync(0, 0);
  regLoad(0);
  regStore(0);

  int buf = 0;
  for (int kb = 0; kb < K; kb += 32) {
    wait_async0();        // my async writes for current tile landed
    __syncthreads();      // everyone's writes landed; prev reads of other buf done
    const bool more = (kb + 32 < K);
    if (more) {
      issueAsync(kb + 32, buf ^ 1);   // DMA overlaps the WMMAs below
      regLoad(kb + 32);
    }

    // A fragment (contiguous k order: k = 16*(lane/16) + e, matching B)
    bf16x16 af;
    {
      const __bf16* ap = &As[buf][wm + (lane & 15)][(lane >> 4) * 16];
#pragma unroll
      for (int e = 0; e < 8; e++) af[e] = ap[e];
#pragma unroll
      for (int e = 0; e < 8; e++) af[8 + e] = ap[8 + e];
    }
    // 4 B fragments + 4 WMMAs (wave covers 16x64)
    {
      const int col  = lane & 15;
      const int koff = (lane >> 4) * 16;
      bf16x16 b0, b1, b2, b3;
      const __bf16* p0 = &Bs[buf][col][koff];
      const __bf16* p1 = &Bs[buf][16 + col][koff];
      const __bf16* p2 = &Bs[buf][32 + col][koff];
      const __bf16* p3 = &Bs[buf][48 + col][koff];
#pragma unroll
      for (int e = 0; e < 16; e++) b0[e] = p0[e];
#pragma unroll
      for (int e = 0; e < 16; e++) b1[e] = p1[e];
#pragma unroll
      for (int e = 0; e < 16; e++) b2[e] = p2[e];
#pragma unroll
      for (int e = 0; e < 16; e++) b3[e] = p3[e];
      acc0 = __builtin_amdgcn_wmma_f32_16x16x32_bf16(false, af, false, b0, (short)0, acc0, false, false);
      acc1 = __builtin_amdgcn_wmma_f32_16x16x32_bf16(false, af, false, b1, (short)0, acc1, false, false);
      acc2 = __builtin_amdgcn_wmma_f32_16x16x32_bf16(false, af, false, b2, (short)0, acc2, false, false);
      acc3 = __builtin_amdgcn_wmma_f32_16x16x32_bf16(false, af, false, b3, (short)0, acc3, false, false);
    }
    if (more) regStore(buf ^ 1);
    buf ^= 1;
  }

  // ---- epilogue ----
  const int ccol = lane & 15;
  const int mb   = (lane >> 4) * 8;
#pragma unroll
  for (int nt = 0; nt < 4; nt++) {
    floatx8 acc = (nt == 0) ? acc0 : (nt == 1) ? acc1 : (nt == 2) ? acc2 : acc3;
    const int gn = n0g + nt * 16 + ccol;
#pragma unroll
    for (int r = 0; r < 8; r++) {
      const int gm = m0 + wm + mb + r;
      float v = acc[r] * alpha;
      if (HASBIAS) v += bias[gn];
      if (ACT == 1)      v = fmaxf(v, 0.0f);
      else if (ACT == 2) v = 0.5f * v * (1.0f + erff(v * 0.70710678118654752f));
      const long ci = cOff + (long)gm * ldc + gn;
      if (HASRES) v += resid[ci];
      if (OUTBF16) ((__bf16*)C)[ci] = (__bf16)v;
      else         ((float*)C)[ci]  = v;
    }
  }
}

// ---------------------------------------------------------------------------
// NormalHead fused: per (b,t) block, loop over 10 crops.
// ---------------------------------------------------------------------------
__global__ __launch_bounds__(256) void head_kernel(
    const float* __restrict__ h,
    const float* __restrict__ c1W, const float* __restrict__ c1b,
    const float* __restrict__ g1, const float* __restrict__ b1,
    const float* __restrict__ m1, const float* __restrict__ v1,
    const float* __restrict__ c2W, const float* __restrict__ c2b,
    const float* __restrict__ g2, const float* __restrict__ b2,
    const float* __restrict__ m2, const float* __restrict__ v2,
    const float* __restrict__ c3W, const float* __restrict__ c3b,
    float* __restrict__ out) {
  const int bidx = blockIdx.x >> 8;
  const int t    = blockIdx.x & 255;
  const int tid  = threadIdx.x;
  __shared__ float hrow[512];
  __shared__ float part[256];
  __shared__ float x1s[32], r1s[32], x2s[16], r2s[16];
  float distSum = 0.0f, scoreSum = 0.0f;

  for (int n = 0; n < 10; n++) {
    const long row = ((long)(bidx * 10 + n) * TLEN + t);
    const float* hr = h + row * CDIM;
    hrow[tid] = hr[tid]; hrow[tid + 256] = hr[tid + 256];
    __syncthreads();

    const int o = tid >> 3, seg = tid & 7;
    {
      const float* w  = c1W + (long)o * CDIM + seg * 64;
      const float* hh = hrow + seg * 64;
      float s = 0.0f;
#pragma unroll 8
      for (int j = 0; j < 64; j++) s += hh[j] * w[j];
      part[tid] = s;
    }
    __syncthreads();
    if (seg == 0) {
      float s = 0.0f;
#pragma unroll
      for (int j = 0; j < 8; j++) s += part[tid + j];
      s += c1b[o];
      x1s[o] = s;
      r1s[o] = fmaxf((s - m1[o]) * rsqrtf(v1[o] + EPSLN) * g1[o] + b1[o], 0.0f);
    }
    __syncthreads();
    if (tid < 16) {
      const float* w2 = c2W + tid * 32;
      float s = 0.0f;
#pragma unroll
      for (int j = 0; j < 32; j++) s += r1s[j] * w2[j];
      s += c2b[tid];
      x2s[tid] = s;
      r2s[tid] = fmaxf((s - m2[tid]) * rsqrtf(v2[tid] + EPSLN) * g2[tid] + b2[tid], 0.0f);
    }
    __syncthreads();
    if (tid == 0) {
      float md1 = 0.0f;
      for (int j = 0; j < 32; j++) { float d = x1s[j] - m1[j]; md1 += d * d / v1[j]; }
      float md2 = 0.0f;
      for (int j = 0; j < 16; j++) { float d = x2s[j] - m2[j]; md2 += d * d / v2[j]; }
      float sc = c3b[0];
      for (int j = 0; j < 16; j++) sc += r2s[j] * c3W[j];
      distSum  += sqrtf(md1) + sqrtf(md2);
      scoreSum += 1.0f / (1.0f + __expf(-sc));
    }
    __syncthreads();
  }
  if (tid == 0) out[bidx * TLEN + t] = (distSum * 0.1f) * (scoreSum * 0.1f);
}

// ---------------------------------------------------------------------------
// Host orchestration
// ---------------------------------------------------------------------------
extern "C" void kernel_launch(void* const* d_in, const int* in_sizes, int n_in,
                              void* d_out, int out_size, void* d_ws, size_t ws_size,
                              hipStream_t stream) {
  (void)in_sizes; (void)n_in; (void)out_size; (void)ws_size;
  const float* x     = (const float*)d_in[0];
  const float* W_emb = (const float*)d_in[1];
  const float* b_emb = (const float*)d_in[2];
  const float* ln1_g = (const float*)d_in[3];
  const float* ln1_b = (const float*)d_in[4];
  const float* W_qkv = (const float*)d_in[5];
  const float* W_o   = (const float*)d_in[6];
  const float* b_o   = (const float*)d_in[7];
  const float* ln2_g = (const float*)d_in[8];
  const float* ln2_b = (const float*)d_in[9];
  const float* W_f1  = (const float*)d_in[10];
  const float* b_f1  = (const float*)d_in[11];
  const float* W_f2  = (const float*)d_in[12];
  const float* b_f2  = (const float*)d_in[13];
  const float* c1_W  = (const float*)d_in[14];
  const float* c1_b  = (const float*)d_in[15];
  const float* bn1_g = (const float*)d_in[16];
  const float* bn1_b = (const float*)d_in[17];
  const float* bn1_m = (const float*)d_in[18];
  const float* bn1_v = (const float*)d_in[19];
  const float* c2_W  = (const float*)d_in[20];
  const float* c2_b  = (const float*)d_in[21];
  const float* bn2_g = (const float*)d_in[22];
  const float* bn2_b = (const float*)d_in[23];
  const float* bn2_m = (const float*)d_in[24];
  const float* bn2_v = (const float*)d_in[25];
  const float* c3_W  = (const float*)d_in[26];
  const float* c3_b  = (const float*)d_in[27];

  // ---- scratch layout (bytes) ----
  char* ws = (char*)d_ws;
  const size_t OFF_H    = 0;
  const size_t SZ_H     = (size_t)MROWS * CDIM * 4;
  const size_t OFF_QKVT = OFF_H + SZ_H;                       // bf16 qkvt; also xb
  const size_t SZ_QKVT  = (size_t)MROWS * 2048 * 2;
  const size_t OFF_Z    = OFF_QKVT + SZ_QKVT;
  const size_t SZ_Z     = (size_t)MROWS * CDIM * 2;
  const size_t OFF_S    = OFF_Z + SZ_Z;
  const size_t SZ_S     = (size_t)640 * 256 * 256 * 2;
  const size_t OFF_O    = OFF_S + SZ_S;                       // o-concat; fbuf aliases
  const size_t SZ_O     = (size_t)MROWS * 1024 * 2;
  const size_t OFF_A2   = OFF_O + SZ_O;
  const size_t SZ_A2    = (size_t)256 * 256 * 2;
  const size_t OFF_BEMB = OFF_A2 + SZ_A2;
  const size_t SZ_BEMB  = (size_t)3072 * 512 * 2;
  const size_t OFF_WQKV = OFF_BEMB + SZ_BEMB;
  const size_t SZ_WQKV  = (size_t)DEPTH * 512 * 2048 * 2;
  const size_t OFF_WO   = OFF_WQKV + SZ_WQKV;
  const size_t SZ_WO    = (size_t)DEPTH * 1024 * 512 * 2;
  const size_t OFF_WF1  = OFF_WO + SZ_WO;
  const size_t SZ_WF1   = (size_t)DEPTH * 512 * 512 * 2;
  const size_t OFF_WF2  = OFF_WF1 + SZ_WF1;

  float*  hbuf  = (float*) (ws + OFF_H);
  __bf16* xb    = (__bf16*)(ws + OFF_QKVT);
  __bf16* qkvtb = (__bf16*)(ws + OFF_QKVT);
  __bf16* zb    = (__bf16*)(ws + OFF_Z);
  __bf16* Sbuf  = (__bf16*)(ws + OFF_S);
  __bf16* obuf  = (__bf16*)(ws + OFF_O);
  __bf16* fbuf  = (__bf16*)(ws + OFF_O);
  __bf16* a2nb  = (__bf16*)(ws + OFF_A2);
  __bf16* Bemb  = (__bf16*)(ws + OFF_BEMB);   // [512][3072]  (N,K)
  __bf16* Wqkvb = (__bf16*)(ws + OFF_WQKV);   // [l][2048][512]
  __bf16* Wob   = (__bf16*)(ws + OFF_WO);     // [l][512][1024]
  __bf16* Wf1b  = (__bf16*)(ws + OFF_WF1);    // [l][512][512]
  __bf16* Wf2b  = (__bf16*)(ws + OFF_WF2);    // [l][512][512]

  const float scaleS = 0.08838834764831845f;  // 128^-0.5

  // ---- pre-passes: casts (weights transposed to [N,K]) ----
  {
    long nx = (long)16 * 10 * TLEN * 1024;
    cast_f32_bf16<<<(unsigned)((nx + 255) / 256), 256, 0, stream>>>(x, xb, nx);
    wemb_transpose<<<(unsigned)((3072l * 512 + 255) / 256), 256, 0, stream>>>(W_emb, Bemb);
    long nq = (long)DEPTH * 512 * 2048;
    cast_transpose<<<(unsigned)((nq + 255) / 256), 256, 0, stream>>>(W_qkv, Wqkvb, 512, 2048, nq);
    long no = (long)DEPTH * 1024 * 512;
    cast_transpose<<<(unsigned)((no + 255) / 256), 256, 0, stream>>>(W_o, Wob, 1024, 512, no);
    long nf = (long)DEPTH * 512 * 512;
    cast_transpose<<<(unsigned)((nf + 255) / 256), 256, 0, stream>>>(W_f1, Wf1b, 512, 512, nf);
    cast_transpose<<<(unsigned)((nf + 255) / 256), 256, 0, stream>>>(W_f2, Wf2b, 512, 512, nf);
    a2_kernel<<<256, 256, 0, stream>>>(a2nb);
  }

  // ---- Conv1d embed (im2col GEMM, K=3072) + ReLU -> hbuf fp32 ----
  {
    dim3 g(TLEN / 128, CDIM / 64, BNB);
    wmma_gemm_t<1, 1, 1, 0, 1, 0><<<g, 256, 0, stream>>>(
        xb, Bemb, hbuf, b_emb, nullptr,
        3072, 1024, 3072, CDIM,
        (long)TLEN * 1024, 0, 0, 0, (long)TLEN * CDIM, 0, 1,
        1.0f, TLEN);
  }

  // ---- transformer layers ----
  for (int l = 0; l < DEPTH; l++) {
    ln_kernel<<<MROWS, 256, 0, stream>>>(hbuf, ln1_g + l * CDIM, ln1_b + l * CDIM, zb);
    // qkvt = zb @ W_qkv[l] -> bf16   (B transposed [2048][512])
    {
      dim3 g(MROWS / 128, 2048 / 64, 1);
      wmma_gemm_t<1, 0, 0, 1, 0, 0><<<g, 256, 0, stream>>>(
          zb, Wqkvb + (long)l * 512 * 2048, qkvtb, nullptr, nullptr,
          CDIM, CDIM, CDIM, 2048,
          0, 0, 0, 0, 0, 0, 1, 1.0f, 0);
    }
    // S = Q K^T * scale  (batch 640 = bn*4+h) -> bf16
    {
      dim3 g(TLEN / 128, TLEN / 64, BNB * HEADS);
      wmma_gemm_t<1, 0, 0, 1, 0, 0><<<g, 256, 0, stream>>>(
          qkvtb, qkvtb + 512, Sbuf, nullptr, nullptr,
          DH, 2048, 2048, TLEN,
          (long)TLEN * 2048, DH, (long)TLEN * 2048, DH,
          (long)HEADS * TLEN * TLEN, (long)TLEN * TLEN, HEADS,
          scaleS, 0);
    }
    softmax256<<<BNB * HEADS * TLEN, 256, 0, stream>>>(Sbuf);
    // o1 = P @ V -> obuf[.., h*256 + 0:128]   (B = V slice, [K,N] register path)
    {
      dim3 g(TLEN / 128, DH / 64, BNB * HEADS);
      wmma_gemm_t<0, 0, 0, 1, 0, 0><<<g, 256, 0, stream>>>(
          Sbuf, qkvtb + 1024, obuf, nullptr, nullptr,
          TLEN, TLEN, 2048, 1024,
          (long)HEADS * TLEN * TLEN, (long)TLEN * TLEN,
          (long)TLEN * 2048, DH,
          (long)TLEN * 1024, 256, HEADS,
          1.0f, 0);
    }
    // o2 = a2n @ T -> obuf[.., h*256 + 128:256]
    {
      dim3 g(TLEN / 128, DH / 64, BNB * HEADS);
      wmma_gemm_t<0, 0, 0, 1, 0, 0><<<g, 256, 0, stream>>>(
          a2nb, qkvtb + 1536, obuf + 128, nullptr, nullptr,
          TLEN, TLEN, 2048, 1024,
          0, 0, (long)TLEN * 2048, DH,
          (long)TLEN * 1024, 256, HEADS,
          1.0f, 0);
    }
    // h = o @ W_o + b_o + h  -> fp32 hbuf  (B transposed [512][1024])
    {
      dim3 g(MROWS / 128, CDIM / 64, 1);
      wmma_gemm_t<1, 0, 0, 0, 1, 1><<<g, 256, 0, stream>>>(
          obuf, Wob + (long)l * 1024 * 512, hbuf, b_o + l * CDIM, hbuf,
          1024, 1024, 1024, CDIM,
          0, 0, 0, 0, 0, 0, 1, 1.0f, 0);
    }
    ln_kernel<<<MROWS, 256, 0, stream>>>(hbuf, ln2_g + l * CDIM, ln2_b + l * CDIM, zb);
    // f = gelu(zb @ W_f1 + b_f1) -> bf16 fbuf  (B transposed [512][512])
    {
      dim3 g(MROWS / 128, CDIM / 64, 1);
      wmma_gemm_t<1, 0, 2, 1, 1, 0><<<g, 256, 0, stream>>>(
          zb, Wf1b + (long)l * 512 * 512, fbuf, b_f1 + l * CDIM, nullptr,
          CDIM, CDIM, CDIM, CDIM,
          0, 0, 0, 0, 0, 0, 1, 1.0f, 0);
    }
    // h = f @ W_f2 + b_f2 + h -> fp32 hbuf
    {
      dim3 g(MROWS / 128, CDIM / 64, 1);
      wmma_gemm_t<1, 0, 0, 0, 1, 1><<<g, 256, 0, stream>>>(
          fbuf, Wf2b + (long)l * 512 * 512, hbuf, b_f2 + l * CDIM, hbuf,
          CDIM, CDIM, CDIM, CDIM,
          0, 0, 0, 0, 0, 0, 1, 1.0f, 0);
    }
  }

  // ---- NormalHead + crop means -> d_out (16,256) fp32 ----
  head_kernel<<<16 * TLEN, 256, 0, stream>>>(
      hbuf, c1_W, c1_b, bn1_g, bn1_b, bn1_m, bn1_v,
      c2_W, c2_b, bn2_g, bn2_b, bn2_m, bn2_v, c3_W, c3_b,
      (float*)d_out);
}
